// ChebNet_35296041238783
// MI455X (gfx1250) — matrix operand order repaired and checked by reading, not compiled, not run on hardware
//
#include <hip/hip_runtime.h>
#include <hip/hip_bf16.h>

typedef __bf16 bf16_t;
typedef __attribute__((ext_vector_type(8)))  __bf16 v8bf;
typedef __attribute__((ext_vector_type(16))) __bf16 v16bf;
typedef __attribute__((ext_vector_type(8)))  float  v8f;

// ---------------------------------------------------------------------------
// Degree / normalization
// ---------------------------------------------------------------------------
__global__ void deg_kernel(const int* __restrict__ row, float* __restrict__ deg, int E) {
    int e = blockIdx.x * blockDim.x + threadIdx.x;
    if (e < E) atomicAdd(&deg[row[e]], 1.0f);
}

__global__ void dinv_kernel(float* __restrict__ d, int n) {
    int i = blockIdx.x * blockDim.x + threadIdx.x;
    if (i < n) {
        float v = d[i];
        d[i] = (v > 0.0f) ? rsqrtf(v) : 0.0f;
    }
}

// ---------------------------------------------------------------------------
// Weight prep: W[K][N] f32 -> Wt[N][K] bf16 (K contiguous per output column)
// ---------------------------------------------------------------------------
__global__ void prep_wt_kernel(const float* __restrict__ W, bf16_t* __restrict__ Wt,
                               int K, int N) {
    int idx = blockIdx.x * blockDim.x + threadIdx.x;
    if (idx < K * N) {
        int n = idx / K;
        int k = idx - n * K;
        Wt[idx] = (bf16_t)W[(size_t)k * N + n];
    }
}

// ---------------------------------------------------------------------------
// Sparse propagation: t[col] += (-dinv[row]*dinv[col]) * h[row]   (h bf16, t f32)
// One wave per edge (grid-stride). Wave-uniform scalars via readfirstlane,
// vectorized b128 gather (8 bf16/lane), f32 atomic scatter (L2-resident t).
// ---------------------------------------------------------------------------
__global__ __launch_bounds__(256)
void prop_kernel(const bf16_t* __restrict__ h, const int* __restrict__ row,
                 const int* __restrict__ col, const float* __restrict__ dinv,
                 float* __restrict__ t, int E, int H) {
    int gwave  = (blockIdx.x * blockDim.x + threadIdx.x) >> 5;
    int lane   = threadIdx.x & 31;
    int nwaves = (gridDim.x * blockDim.x) >> 5;
    for (int e = gwave; e < E; e += nwaves) {
        int r = __builtin_amdgcn_readfirstlane(row[e]);
        int c = __builtin_amdgcn_readfirstlane(col[e]);
        float norm = -dinv[r] * dinv[c];
        if (norm != 0.0f) {
            const bf16_t* hr = h + (size_t)r * H;
            float*        tc = t + (size_t)c * H;
            #pragma unroll
            for (int j = 0; j < 2; ++j) {            // H = 512 = 2 * 32 * 8
                const int f = (lane + 32 * j) * 8;
                v8bf hv = *(const v8bf*)(hr + f);
                #pragma unroll
                for (int q = 0; q < 8; ++q)
                    atomicAdd(&tc[f + q], norm * (float)hv[q]);
            }
        }
    }
}

// ---------------------------------------------------------------------------
// WMMA GEMM:  C = act( A @ W  [+ A2 @ W2]  + bias )
//   A  : [M x K]  bf16 (or f32 if A_IS_F32), row-major
//   A2 : [M x K]  f32 (only when DUAL)
//   Wt : [N x K]  bf16 (pre-transposed, K contiguous)
//   C  : [M x N]  bf16 or f32
// Block: 64x64 tile, BK=32, 256 threads = 8 waves (2 x 4), 2 subtiles/wave.
// ---------------------------------------------------------------------------
constexpr int BM = 64, BN = 64, BK = 32;
constexpr int LSTR = BK + 8;   // LDS row stride in bf16 elems (80B, conflict-free b128)

template<bool A_IS_F32, bool DUAL, bool SILU, bool OUT_BF16>
__global__ __launch_bounds__(256)
void gemm_wmma(const void* __restrict__ Ap, const void* __restrict__ A2p,
               const bf16_t* __restrict__ Wt, const bf16_t* __restrict__ W2t,
               const float* __restrict__ bias, void* __restrict__ Cp,
               int M, int K, int N) {
    __shared__ bf16_t As [BM * LSTR];
    __shared__ bf16_t Bs [BN * LSTR];
    __shared__ bf16_t As2[DUAL ? BM * LSTR : 8];
    __shared__ bf16_t Bs2[DUAL ? BN * LSTR : 8];

    const int m0   = blockIdx.x * BM;
    const int n0   = blockIdx.y * BN;
    const int tid  = threadIdx.x;
    const int lane = tid & 31;
    const int w    = tid >> 5;
    const int wm   = w >> 2;        // 0..1
    const int wn   = w & 3;         // 0..3
    const int lr   = lane & 15;
    const int half = lane >> 4;

    const int ldr = tid >> 2;       // 0..63  tile row loaded by this thread
    const int ldc = (tid & 3) * 8;  // 0/8/16/24

    v8f acc0 = {};
    v8f acc1 = {};

    for (int k0 = 0; k0 < K; k0 += BK) {
        // ---- stage A tile ----
        {
            const int gr = m0 + ldr;
            v8bf av = {};
            if constexpr (A_IS_F32) {
                if (gr < M) {
                    const float* src = (const float*)Ap + (size_t)gr * K + k0 + ldc;
                    float4 f0 = *(const float4*)(src);
                    float4 f1 = *(const float4*)(src + 4);
                    av[0] = (bf16_t)f0.x; av[1] = (bf16_t)f0.y;
                    av[2] = (bf16_t)f0.z; av[3] = (bf16_t)f0.w;
                    av[4] = (bf16_t)f1.x; av[5] = (bf16_t)f1.y;
                    av[6] = (bf16_t)f1.z; av[7] = (bf16_t)f1.w;
                }
            } else {
                if (gr < M)
                    av = *(const v8bf*)((const bf16_t*)Ap + (size_t)gr * K + k0 + ldc);
            }
            *(v8bf*)&As[ldr * LSTR + ldc] = av;
        }
        if constexpr (DUAL) {
            const int gr = m0 + ldr;
            v8bf av = {};
            if (gr < M) {
                const float* src = (const float*)A2p + (size_t)gr * K + k0 + ldc;
                float4 f0 = *(const float4*)(src);
                float4 f1 = *(const float4*)(src + 4);
                av[0] = (bf16_t)f0.x; av[1] = (bf16_t)f0.y;
                av[2] = (bf16_t)f0.z; av[3] = (bf16_t)f0.w;
                av[4] = (bf16_t)f1.x; av[5] = (bf16_t)f1.y;
                av[6] = (bf16_t)f1.z; av[7] = (bf16_t)f1.w;
            }
            *(v8bf*)&As2[ldr * LSTR + ldc] = av;
        }
        // ---- stage B tile(s) (N,K always multiples of tile; no guard) ----
        {
            v8bf bv = *(const v8bf*)(Wt + (size_t)(n0 + ldr) * K + k0 + ldc);
            *(v8bf*)&Bs[ldr * LSTR + ldc] = bv;
            if constexpr (DUAL) {
                v8bf b2 = *(const v8bf*)(W2t + (size_t)(n0 + ldr) * K + k0 + ldc);
                *(v8bf*)&Bs2[ldr * LSTR + ldc] = b2;
            }
        }
        __syncthreads();

        // ---- B fragment: lane = col lr, lo8 @ K=half*8, hi8 @ K=16+half*8 ----
        {
            const bf16_t* bp = &Bs[(wn * 16 + lr) * LSTR];
            v8bf blo = *(const v8bf*)(bp + half * 8);
            v8bf bhi = *(const v8bf*)(bp + 16 + half * 8);
            v16bf bf = __builtin_shufflevector(blo, bhi,
                        0,1,2,3,4,5,6,7,8,9,10,11,12,13,14,15);

            // ---- A fragments: row = lr, lo8 @ K=half*8, hi8 @ K=16+half*8 ----
            const bf16_t* ap0 = &As[(wm * 32 + lr) * LSTR];
            v8bf a0l = *(const v8bf*)(ap0 + half * 8);
            v8bf a0h = *(const v8bf*)(ap0 + 16 + half * 8);
            v16bf a0 = __builtin_shufflevector(a0l, a0h,
                        0,1,2,3,4,5,6,7,8,9,10,11,12,13,14,15);
            acc0 = __builtin_amdgcn_wmma_f32_16x16x32_bf16(
                       false, a0, false, bf, (short)0, acc0, false, false);

            const bf16_t* ap1 = &As[(wm * 32 + 16 + lr) * LSTR];
            v8bf a1l = *(const v8bf*)(ap1 + half * 8);
            v8bf a1h = *(const v8bf*)(ap1 + 16 + half * 8);
            v16bf a1 = __builtin_shufflevector(a1l, a1h,
                        0,1,2,3,4,5,6,7,8,9,10,11,12,13,14,15);
            acc1 = __builtin_amdgcn_wmma_f32_16x16x32_bf16(
                       false, a1, false, bf, (short)0, acc1, false, false);
        }
        if constexpr (DUAL) {
            const bf16_t* bp = &Bs2[(wn * 16 + lr) * LSTR];
            v8bf blo = *(const v8bf*)(bp + half * 8);
            v8bf bhi = *(const v8bf*)(bp + 16 + half * 8);
            v16bf bf = __builtin_shufflevector(blo, bhi,
                        0,1,2,3,4,5,6,7,8,9,10,11,12,13,14,15);

            const bf16_t* ap0 = &As2[(wm * 32 + lr) * LSTR];
            v8bf a0l = *(const v8bf*)(ap0 + half * 8);
            v8bf a0h = *(const v8bf*)(ap0 + 16 + half * 8);
            v16bf a0 = __builtin_shufflevector(a0l, a0h,
                        0,1,2,3,4,5,6,7,8,9,10,11,12,13,14,15);
            acc0 = __builtin_amdgcn_wmma_f32_16x16x32_bf16(
                       false, a0, false, bf, (short)0, acc0, false, false);

            const bf16_t* ap1 = &As2[(wm * 32 + 16 + lr) * LSTR];
            v8bf a1l = *(const v8bf*)(ap1 + half * 8);
            v8bf a1h = *(const v8bf*)(ap1 + 16 + half * 8);
            v16bf a1 = __builtin_shufflevector(a1l, a1h,
                        0,1,2,3,4,5,6,7,8,9,10,11,12,13,14,15);
            acc1 = __builtin_amdgcn_wmma_f32_16x16x32_bf16(
                       false, a1, false, bf, (short)0, acc1, false, false);
        }
        __syncthreads();
    }

    // ---- epilogue: D layout -> M = subtile + half*8 + j, N = lane&15 ----
    const int   nn = n0 + wn * 16 + lr;
    const float bv = bias[nn];
    #pragma unroll
    for (int s = 0; s < 2; ++s) {
        v8f acc = s ? acc1 : acc0;
        const int mbase = m0 + wm * 32 + s * 16 + half * 8;
        #pragma unroll
        for (int j = 0; j < 8; ++j) {
            const int m = mbase + j;
            if (m < M) {
                float v = acc[j] + bv;
                if constexpr (SILU) v = v / (1.0f + __expf(-v));
                if constexpr (OUT_BF16)
                    ((bf16_t*)Cp)[(size_t)m * N + nn] = (bf16_t)v;
                else
                    ((float*)Cp)[(size_t)m * N + nn] = v;
            }
        }
    }
}

// ---------------------------------------------------------------------------
// Launch
// ---------------------------------------------------------------------------
extern "C" void kernel_launch(void* const* d_in, const int* in_sizes, int n_in,
                              void* d_out, int out_size, void* d_ws, size_t ws_size,
                              hipStream_t stream) {
    (void)in_sizes; (void)n_in; (void)out_size; (void)ws_size;

    constexpr int NN = 50000, E = 400000;
    constexpr int IN = 256, H = 512, OUT = 256;

    const float* x      = (const float*)d_in[0];
    const int*   ei     = (const int*)  d_in[1];   // [2][E]
    const float* in_w   = (const float*)d_in[2];   // [IN][H]
    const float* in_b   = (const float*)d_in[3];   // [H]
    const float* conv_w = (const float*)d_in[4];   // [2][2][H][H]
    const float* conv_b = (const float*)d_in[5];   // [2][H]
    const float* out_w  = (const float*)d_in[6];   // [H][OUT]
    const float* out_b  = (const float*)d_in[7];   // [OUT]
    float*       y      = (float*)d_out;           // [NN][OUT]

    const int* row = ei;
    const int* col = ei + E;

    // ---- carve workspace ----
    char* p = (char*)d_ws;
    auto carve = [&](size_t bytes) -> void* {
        void* r = (void*)p;
        p += (bytes + 255) & ~(size_t)255;
        return r;
    };
    float*  dinv   = (float*) carve((size_t)NN * 4);
    bf16_t* in_wt  = (bf16_t*)carve((size_t)IN * H * 2);
    bf16_t* cw[4];
    for (int i = 0; i < 4; ++i) cw[i] = (bf16_t*)carve((size_t)H * H * 2);
    bf16_t* out_wt = (bf16_t*)carve((size_t)H * OUT * 2);
    bf16_t* hA     = (bf16_t*)carve((size_t)NN * H * 2);
    bf16_t* hB     = (bf16_t*)carve((size_t)NN * H * 2);
    float*  t      = (float*) carve((size_t)NN * H * 4);

    // ---- degree / dinv ----
    hipMemsetAsync(dinv, 0, (size_t)NN * 4, stream);
    deg_kernel <<<(E  + 255) / 256, 256, 0, stream>>>(row, dinv, E);
    dinv_kernel<<<(NN + 255) / 256, 256, 0, stream>>>(dinv, NN);

    // ---- weight prep (f32 [K][N] -> bf16 [N][K]) ----
    prep_wt_kernel<<<(IN * H + 255) / 256, 256, 0, stream>>>(in_w, in_wt, IN, H);
    for (int i = 0; i < 4; ++i)
        prep_wt_kernel<<<(H * H + 255) / 256, 256, 0, stream>>>(
            conv_w + (size_t)i * H * H, cw[i], H, H);
    prep_wt_kernel<<<(H * OUT + 255) / 256, 256, 0, stream>>>(out_w, out_wt, H, OUT);

    const dim3 blk(256);
    const dim3 g_in ((NN + BM - 1) / BM, H   / BN);
    const dim3 g_cnv((NN + BM - 1) / BM, H   / BN);
    const dim3 g_out((NN + BM - 1) / BM, OUT / BN);

    // ---- hA = silu(x @ in_w + in_b) ----
    gemm_wmma<true, false, true, true><<<g_in, blk, 0, stream>>>(
        (const void*)x, nullptr, in_wt, nullptr, in_b, (void*)hA, NN, IN, H);

    // ---- layer 0: t = prop(hA);  hB = silu(hA@W00 + t@W01 + b0) ----
    hipMemsetAsync(t, 0, (size_t)NN * H * 4, stream);
    prop_kernel<<<2048, 256, 0, stream>>>(hA, row, col, dinv, t, E, H);
    gemm_wmma<false, true, true, true><<<g_cnv, blk, 0, stream>>>(
        (const void*)hA, (const void*)t, cw[0], cw[1], conv_b, (void*)hB, NN, H, H);

    // ---- layer 1: t = prop(hB);  hA = silu(hB@W10 + t@W11 + b1) ----
    hipMemsetAsync(t, 0, (size_t)NN * H * 4, stream);
    prop_kernel<<<2048, 256, 0, stream>>>(hB, row, col, dinv, t, E, H);
    gemm_wmma<false, true, true, true><<<g_cnv, blk, 0, stream>>>(
        (const void*)hB, (const void*)t, cw[2], cw[3], conv_b + H, (void*)hA, NN, H, H);

    // ---- y = hA @ out_w + out_b ----
    gemm_wmma<false, false, false, false><<<g_out, blk, 0, stream>>>(
        (const void*)hA, nullptr, out_wt, nullptr, out_b, (void*)y, NN, H, OUT);
}
